// CrossAttention_23845658428179
// MI455X (gfx1250) — matrix operand order repaired and checked by reading, not compiled, not run on hardware
//
#include <hip/hip_runtime.h>

// CrossAttention for MI455X (gfx1250, wave32, WMMA).
// ~55 GFLOP over ~140MB -> ~390 FLOP/byte: compute bound -> bf16 WMMA path.
// All operands pre-swizzled into fragment-native order so every WMMA fragment
// is two b128 loads. Attention K/V staged into LDS via the Tensor Data Mover.

#define HEADS 8
#define SCALE 0.125f  // 64^-0.5

typedef __attribute__((ext_vector_type(16))) __bf16 v16bf;
typedef __attribute__((ext_vector_type(2)))  __bf16 v2bf;
typedef __attribute__((ext_vector_type(8)))  float  v8f;
typedef __attribute__((ext_vector_type(4)))  unsigned int u32x4;
typedef __attribute__((ext_vector_type(8)))  int i32x8;
typedef __attribute__((ext_vector_type(4)))  int i32x4;

union AFrag {
  unsigned short u[16];
  unsigned int   u32[8];
  uint4          q[2];
  v16bf          v;
};

__device__ __forceinline__ unsigned short f2bf(float f) {
  return __builtin_bit_cast(unsigned short, (__bf16)f);
}
__device__ __forceinline__ unsigned int pk2(float a, float b) {
  v2bf r = {(__bf16)a, (__bf16)b};
  return __builtin_bit_cast(unsigned int, r);
}
__device__ __forceinline__ uint4 pk8(v8f v) {
  return make_uint4(pk2(v[0], v[1]), pk2(v[2], v[3]), pk2(v[4], v[5]), pk2(v[6], v[7]));
}

// Fragment-native swizzles (ISA 7.12.2, 16-bit A 16x32 / B 32x16, wave32).
__device__ __forceinline__ int aswz(int m, int k) {
  return ((m >> 4) * 32 + (m & 15) + 16 * ((k >> 3) & 1)) * 16 + ((k >> 4) * 8 + (k & 7));
}
__device__ __forceinline__ int bswz(int k, int n) {
  return ((n >> 4) * 32 + (n & 15) + 16 * ((k >> 3) & 1)) * 16 + ((k >> 4) * 8 + (k & 7));
}

__device__ __forceinline__ v16bf ld_frag(const unsigned short* p) {
  AFrag f;
  f.q[0] = *(const uint4*)p;
  f.q[1] = *(const uint4*)(p + 8);
  return f.v;
}

__device__ __forceinline__ v8f wmma_bf16(v16bf a, v16bf b, v8f c) {
  return __builtin_amdgcn_wmma_f32_16x16x32_bf16(false, a, false, b,
                                                 (short)0, c, false, false);
}

// TDM: contiguous 32KB global -> LDS copy as a 1-row tensor of 4096 x 8B.
// D# per CDNA5 ISA ch.8: group0 {count=1, lds_addr, global_addr, type=2},
// group1 {data_size=3(8B), tensor_dim0=4096, tensor_dim1=1, tile_dim0=4096,
//         tile_dim1=1, tensor_dim0_stride=4096}.
__device__ __forceinline__ void tdm_copy_32k(unsigned lds_off, const void* gptr) {
  unsigned long long ga = (unsigned long long)gptr;
  u32x4 g0;
  g0[0] = 1u;                                               // count=1, user D#
  g0[1] = lds_off;                                          // lds_addr (bytes)
  g0[2] = (unsigned)(ga & 0xFFFFFFFFu);                     // global_addr[31:0]
  g0[3] = (unsigned)((ga >> 32) & 0x1FFFFFFu) | (2u << 30); // [56:32] | type=2
  i32x8 g1;
  g1[0] = 3 << 16;                 // data_size = 8B
  g1[1] = (int)(4096u << 16);      // tensor_dim0[15:0] in bits 63:48
  g1[2] = 1 << 16;                 // tensor_dim0 hi=0 | tensor_dim1 lo = 1
  g1[3] = (int)(4096u << 16);      // tensor_dim1 hi=0 | tile_dim0 = 4096
  g1[4] = 1;                       // tile_dim1 = 1, tile_dim2 = 0
  g1[5] = 4096;                    // tensor_dim0_stride lo32
  g1[6] = 0;
  g1[7] = 0;
  i32x4 z = {0, 0, 0, 0};
#if __clang_major__ >= 23
  i32x8 z8 = {0, 0, 0, 0, 0, 0, 0, 0};
  __builtin_amdgcn_tensor_load_to_lds(g0, g1, z, z, z8, 0);
#else
  __builtin_amdgcn_tensor_load_to_lds(g0, g1, z, z, 0);
#endif
}

// ---------------------------------------------------------------------------
// Kernel 1: Q projection. Q[e,p] = sum_c Wq[e,c]*X[b,c,p], *SCALE, bf16.
// ---------------------------------------------------------------------------
__global__ __launch_bounds__(256) void proj_q_kernel(
    const float* __restrict__ x, const float* __restrict__ Wq,
    unsigned short* __restrict__ q_ws) {
  const int b  = blockIdx.z;
  const int m0 = blockIdx.y * 64;   // e
  const int n0 = blockIdx.x * 64;   // p
  __shared__ uint4 As4[256];
  __shared__ uint4 Bs4[256];
  unsigned short* As = (unsigned short*)As4;
  unsigned short* Bs = (unsigned short*)Bs4;
  const int t = threadIdx.x, w = t >> 5, lane = t & 31;
  const int ln = lane & 15, hf = lane >> 4;
  const int tm = w & 3, tn2 = w >> 2;
  v8f acc0 = {}, acc1 = {};
  const float* Xb = x + (size_t)b * 512 * 4096;
  for (int k0 = 0; k0 < 512; k0 += 32) {
    __syncthreads();
#pragma unroll
    for (int i = 0; i < 4; ++i) {            // A: 64x32, pairs along k
      int pi = i * 256 + t;
      int row = pi >> 4, col = (pi & 15) * 2;
      float2 v = *(const float2*)&Wq[(size_t)(m0 + row) * 512 + (k0 + col)];
      *(unsigned int*)&As[aswz(row, col)] = pk2(v.x, v.y);
    }
#pragma unroll
    for (int i = 0; i < 4; ++i) {            // B: 32x64, pairs along n (coalesced)
      int pi = i * 256 + t;
      int row = pi >> 5, col = (pi & 31) * 2;
      float2 v = *(const float2*)&Xb[(size_t)(k0 + row) * 4096 + (n0 + col)];
      Bs[bswz(row, col)]     = f2bf(v.x);
      Bs[bswz(row, col + 1)] = f2bf(v.y);
    }
    __syncthreads();
    v16bf a  = ld_frag(As + (tm * 32 + lane) * 16);
    v16bf b0 = ld_frag(Bs + ((tn2 * 2 + 0) * 32 + lane) * 16);
    v16bf b1 = ld_frag(Bs + ((tn2 * 2 + 1) * 32 + lane) * 16);
    acc0 = wmma_bf16(a, b0, acc0);
    acc1 = wmma_bf16(a, b1, acc1);
  }
  int base_e = m0 + tm * 16 + 8 * hf;        // channel base (mult of 8)
  int head = base_e >> 6, d0 = base_e & 63;
  int s = d0 >> 5, hfq = (d0 >> 3) & 1, e0 = ((d0 >> 4) & 1) * 8;
  size_t bhb = (size_t)(b * HEADS + head) * 4096 * 64;
  int p0 = n0 + tn2 * 32 + ln;
  *(uint4*)&q_ws[bhb + (((size_t)p0 * 2 + s) * 2 + hfq) * 16 + e0]        = pk8(acc0 * SCALE);
  *(uint4*)&q_ws[bhb + (((size_t)(p0 + 16) * 2 + s) * 2 + hfq) * 16 + e0] = pk8(acc1 * SCALE);
}

// ---------------------------------------------------------------------------
// Kernel 2: KV projection. e2<512 -> K (B-swizzled over d), else V (over tok).
// ---------------------------------------------------------------------------
__global__ __launch_bounds__(256) void proj_kv_kernel(
    const float* __restrict__ ctx, const float* __restrict__ Wkv,
    unsigned short* __restrict__ k_ws, unsigned short* __restrict__ v_ws) {
  const int b  = blockIdx.z;
  const int m0 = blockIdx.y * 64;   // e2
  const int n0 = blockIdx.x * 64;   // tok
  __shared__ uint4 As4[256];
  __shared__ uint4 Bs4[256];
  unsigned short* As = (unsigned short*)As4;
  unsigned short* Bs = (unsigned short*)Bs4;
  const int t = threadIdx.x, w = t >> 5, lane = t & 31;
  const int ln = lane & 15, hf = lane >> 4;
  const int tm = w & 3, tn2 = w >> 2;
  v8f acc0 = {}, acc1 = {};
  for (int k0 = 0; k0 < 768; k0 += 32) {
    __syncthreads();
#pragma unroll
    for (int i = 0; i < 4; ++i) {            // A: Wkv 64x32 pairs along k
      int pi = i * 256 + t;
      int row = pi >> 4, col = (pi & 15) * 2;
      float2 v = *(const float2*)&Wkv[(size_t)(m0 + row) * 768 + (k0 + col)];
      *(unsigned int*)&As[aswz(row, col)] = pk2(v.x, v.y);
    }
#pragma unroll
    for (int i = 0; i < 4; ++i) {            // B: ctx^T 32x64, pairs along k
      int pi = i * 256 + t;
      int col = pi >> 4, row = (pi & 15) * 2;
      float2 v = *(const float2*)&ctx[((size_t)b * 256 + (n0 + col)) * 768 + (k0 + row)];
      *(unsigned int*)&Bs[bswz(row, col)] = pk2(v.x, v.y);
    }
    __syncthreads();
    v16bf a  = ld_frag(As + (tm * 32 + lane) * 16);
    v16bf b0 = ld_frag(Bs + ((tn2 * 2 + 0) * 32 + lane) * 16);
    v16bf b1 = ld_frag(Bs + ((tn2 * 2 + 1) * 32 + lane) * 16);
    acc0 = wmma_bf16(a, b0, acc0);
    acc1 = wmma_bf16(a, b1, acc1);
  }
  int e2b = m0 + tm * 16 + 8 * hf;           // channel base (mult of 8)
  int tok0 = n0 + tn2 * 32 + ln;
  if (e2b < 512) {                           // K: lane rows contiguous -> b128
    int head = e2b >> 6, d0 = e2b & 63;
    int s = d0 >> 5, hfL = (d0 >> 3) & 1, e0 = ((d0 >> 4) & 1) * 8;
    size_t base = (size_t)(b * HEADS + head) * 16384;
    int j_ = tok0 & 15;
    *(uint4*)&k_ws[base + (((tok0 >> 4) * 2 + s) * 32 + j_ + 16 * hfL) * 16 + e0] = pk8(acc0);
    *(uint4*)&k_ws[base + ((((tok0 + 16) >> 4) * 2 + s) * 32 + j_ + 16 * hfL) * 16 + e0] = pk8(acc1);
  } else {                                   // V: transpose -> scatter b16
    int e3 = e2b - 512;
    int head = e3 >> 6, d0 = e3 & 63;
    size_t base = (size_t)(b * HEADS + head) * 16384;
#pragma unroll
    for (int which = 0; which < 2; ++which) {
      int tok = tok0 + which * 16;
      int sv = tok >> 5, hfv = (tok >> 3) & 1, ev = ((tok >> 4) & 1) * 8 + (tok & 7);
      v8f ac = which ? acc1 : acc0;
#pragma unroll
      for (int r = 0; r < 8; ++r) {
        int d = d0 + r;
        v_ws[base + ((sv * 4 + (d >> 4)) * 32 + (d & 15) + 16 * hfv) * 16 + ev] = f2bf(ac[r]);
      }
    }
  }
}

// ---------------------------------------------------------------------------
// Kernel 3: attention per (b,head): 128 query rows per block, 8 waves x 16.
// K/V staged into LDS by the Tensor Data Mover; sim = q.k^T; softmax; P.V.
// ---------------------------------------------------------------------------
__global__ __launch_bounds__(256) void attn_kernel(
    const unsigned short* __restrict__ q_ws, const unsigned short* __restrict__ k_ws,
    const unsigned short* __restrict__ v_ws, unsigned short* __restrict__ az) {
  const int bh = blockIdx.y;
  const int p0 = blockIdx.x * 128;
  __shared__ uint4 Ks4[2048];   // 32KB: K fragments
  __shared__ uint4 Vs4[2048];   // 32KB: V fragments
  __shared__ uint4 Ps4[4096];   // 64KB: per-wave P staging (8 x 8KB)
  const int t = threadIdx.x, w = t >> 5, lane = t & 31;
  const int ln = lane & 15, hf = lane >> 4;

  if (w == 0) {                  // one wave drives both TDM transfers
    tdm_copy_32k((unsigned)(size_t)(void*)&Ks4[0], k_ws + (size_t)bh * 16384);
    tdm_copy_32k((unsigned)(size_t)(void*)&Vs4[0], v_ws + (size_t)bh * 16384);
    __builtin_amdgcn_s_wait_tensorcnt(0);
  }
  __syncthreads();
  const unsigned short* Ks = (const unsigned short*)Ks4;
  const unsigned short* Vs = (const unsigned short*)Vs4;
  unsigned short* Ps = (unsigned short*)Ps4;

  const int prow = p0 + w * 16;
  const unsigned short* qb = q_ws + (size_t)bh * 4096 * 64;
  AFrag qa[2];
#pragma unroll
  for (int s = 0; s < 2; ++s) {
    const uint4* src = (const uint4*)(qb + (((size_t)(prow + ln) * 2 + s) * 2 + hf) * 16);
    qa[s].q[0] = src[0];
    qa[s].q[1] = src[1];
  }

  v8f sim[16];
#pragma unroll
  for (int jt = 0; jt < 16; ++jt) {
    v8f acc = {};
#pragma unroll
    for (int s = 0; s < 2; ++s)
      acc = wmma_bf16(qa[s].v, ld_frag(Ks + ((jt * 2 + s) * 32 + lane) * 16), acc);
    sim[jt] = acc;
  }

  // softmax over 256 cols; row m=r+8*hf lives in one 16-lane half-group
  float rmax[8], rinv[8], rsum[8];
#pragma unroll
  for (int r = 0; r < 8; ++r) {
    float m = sim[0][r];
#pragma unroll
    for (int jt = 1; jt < 16; ++jt) m = fmaxf(m, sim[jt][r]);
#pragma unroll
    for (int off = 1; off < 16; off <<= 1) m = fmaxf(m, __shfl_xor(m, off, 32));
    rmax[r] = m;
    rsum[r] = 0.f;
  }
#pragma unroll
  for (int jt = 0; jt < 16; ++jt)
#pragma unroll
    for (int r = 0; r < 8; ++r) {
      float e = __expf(sim[jt][r] - rmax[r]);
      sim[jt][r] = e;
      rsum[r] += e;
    }
#pragma unroll
  for (int r = 0; r < 8; ++r) {
    float s = rsum[r];
#pragma unroll
    for (int off = 1; off < 16; off <<= 1) s += __shfl_xor(s, off, 32);
    rinv[r] = 1.0f / s;
  }

  // transpose P: D-layout -> A-fragment layout (scatter b16, wave-private)
#pragma unroll
  for (int jt = 0; jt < 16; ++jt)
#pragma unroll
    for (int r = 0; r < 8; ++r) {
      int m = r + 8 * hf;
      int j = jt * 16 + ln;
      int s = j >> 5, rem = j & 31;
      Ps[(w << 12) + ((s * 16 + m) * 2 + ((rem >> 3) & 1)) * 16 +
         ((rem >> 4) * 8 + (rem & 7))] = f2bf(sim[jt][r]);
    }

  v8f o[4] = {};
#pragma unroll
  for (int s = 0; s < 8; ++s) {
    v16bf pa = ld_frag(Ps + (w << 12) + ((s * 16 + ln) * 2 + hf) * 16);
#pragma unroll
    for (int dt = 0; dt < 4; ++dt)
      o[dt] = wmma_bf16(pa, ld_frag(Vs + ((s * 4 + dt) * 32 + lane) * 16), o[dt]);
  }

  // epilogue: write az in tile-swizzled order (per-batch, 2048-ushort tiles)
  const int b = bh >> 3, head = bh & 7;
  unsigned short* azb = az + (size_t)b * 512 * 4096;
#pragma unroll
  for (int dt = 0; dt < 4; ++dt) {
    int e = head * 64 + dt * 16 + ln;
    int ks = e >> 5, ke = e & 31;
#pragma unroll
    for (int r = 0; r < 8; ++r) {
      int p = prow + r + 8 * hf;
      azb[(size_t)(ks * 64 + (p >> 6)) * 2048 + bswz(ke, p & 63)] =
          f2bf(o[dt][r] * rinv[r]);
    }
  }
}

// ---------------------------------------------------------------------------
// Kernel 4: out projection + bias + residual.
// ---------------------------------------------------------------------------
__global__ __launch_bounds__(256) void proj_out_kernel(
    const unsigned short* __restrict__ az, const float* __restrict__ Wout,
    const float* __restrict__ bout, const float* __restrict__ x,
    float* __restrict__ out) {
  const int b  = blockIdx.z;
  const int m0 = blockIdx.y * 64;   // c
  const int n0 = blockIdx.x * 64;   // p
  __shared__ uint4 As4[256];
  __shared__ uint4 Bs4[256];
  unsigned short* As = (unsigned short*)As4;
  unsigned short* Bs = (unsigned short*)Bs4;
  const int t = threadIdx.x, w = t >> 5, lane = t & 31;
  const int ln = lane & 15, hf = lane >> 4;
  const int tm = w & 3, tn2 = w >> 2;
  v8f acc0 = {}, acc1 = {};
  const unsigned short* Ab = az + (size_t)b * 512 * 4096;
  for (int k0 = 0; k0 < 512; k0 += 32) {
    __syncthreads();
#pragma unroll
    for (int i = 0; i < 4; ++i) {
      int pi = i * 256 + t;
      int row = pi >> 4, col = (pi & 15) * 2;
      float2 v = *(const float2*)&Wout[(size_t)(m0 + row) * 512 + (k0 + col)];
      *(unsigned int*)&As[aswz(row, col)] = pk2(v.x, v.y);
    }
    Bs4[t] = ((const uint4*)(Ab + (size_t)((k0 >> 5) * 64 + (n0 >> 6)) * 2048))[t];
    __syncthreads();
    v16bf a  = ld_frag(As + (tm * 32 + lane) * 16);
    v16bf b0 = ld_frag(Bs + ((tn2 * 2 + 0) * 32 + lane) * 16);
    v16bf b1 = ld_frag(Bs + ((tn2 * 2 + 1) * 32 + lane) * 16);
    acc0 = wmma_bf16(a, b0, acc0);
    acc1 = wmma_bf16(a, b1, acc1);
  }
#pragma unroll
  for (int r = 0; r < 8; ++r) {
    int c = m0 + tm * 16 + r + 8 * hf;
    int p = n0 + tn2 * 32 + ln;
    size_t i0 = ((size_t)b * 512 + c) * 4096 + p;
    out[i0]      = acc0[r] + bout[c] + x[i0];
    out[i0 + 16] = acc1[r] + bout[c] + x[i0 + 16];
  }
}

// ---------------------------------------------------------------------------
extern "C" void kernel_launch(void* const* d_in, const int* in_sizes, int n_in,
                              void* d_out, int out_size, void* d_ws, size_t ws_size,
                              hipStream_t stream) {
  const float* x    = (const float*)d_in[0];
  const float* ctx  = (const float*)d_in[1];
  const float* Wq   = (const float*)d_in[2];
  const float* Wkv  = (const float*)d_in[3];
  const float* Wout = (const float*)d_in[4];
  const float* bout = (const float*)d_in[5];
  float* out = (float*)d_out;

  unsigned short* q_ws = (unsigned short*)d_ws;             // 8*512*4096
  unsigned short* k_ws = q_ws + (size_t)8 * 512 * 4096;     // 64*16384
  unsigned short* v_ws = k_ws + (size_t)64 * 16384;
  unsigned short* az   = v_ws + (size_t)64 * 16384;         // 8*512*4096

  proj_q_kernel<<<dim3(64, 8, 8), 256, 0, stream>>>(x, Wq, q_ws);
  proj_kv_kernel<<<dim3(4, 16, 8), 256, 0, stream>>>(ctx, Wkv, k_ws, v_ws);
  attn_kernel<<<dim3(32, 64), 256, 0, stream>>>(q_ws, k_ws, v_ws, az);
  proj_out_kernel<<<dim3(64, 8, 8), 256, 0, stream>>>(az, Wout, bout, x, out);
}
